// BDHHRM_23424751632641
// MI455X (gfx1250) — matrix operand order repaired and checked by reading, not compiled, not run on hardware
//
#include <hip/hip_runtime.h>
#include <hip/hip_bf16.h>
#include <math.h>

// ---------------------------------------------------------------------------
// HRM forward for MI455X (gfx1250, wave32, WMMA).
// All GEMMs run as bf16 x bf16 -> f32 via v_wmma_f32_16x16x32_bf16.
// GEMM core: 128x64 workgroup tile, 32x32 per wave (4 WMMA/K-step),
// register-buffered software pipeline over K. B-staging path selected at
// compile time via template (contiguous vs generic-stride/QR^T).
// ---------------------------------------------------------------------------

typedef __attribute__((ext_vector_type(16))) __bf16 v16bf;
typedef __attribute__((ext_vector_type(8)))  float  v8f;

#define WG  256
#define BM  128
#define BN  64
#define BK  32
#define PAD 4     // pad rows to 72B: keeps b64 LDS stores aligned, spreads banks

#define EPI_NONE      0
#define EPI_RELU      1
#define EPI_MASKSCALE 2

// ---------------------------------------------------------------------------
// Generic batched GEMM:  C[b,h] = A[b,h] (MxK, row-major, ld=K)
//                              x  B[b,h] (addressed as Bptr + k*sBk + n*sBn)
// CONTIG: B is row-major with sBn==1 (vectorized staging).
// ---------------------------------------------------------------------------
template <bool CONTIG>
__global__ __launch_bounds__(WG) void k_gemm(
    const __bf16* __restrict__ A, long sAb, long sAh,
    const __bf16* __restrict__ B, long sBb, long sBh, long sBk, long sBn,
    float* __restrict__ Cf, __bf16* __restrict__ Cb, long sCb, long sCh,
    int M, int Nd, int K, int nh, int epi, float scale)
{
    __shared__ __bf16 As[BM][BK + PAD];   // row-major, K contiguous (9216 B)
    __shared__ __bf16 Bs[BN][BK + PAD];   // transposed into LDS: Bs[n][k]

    const int bi = blockIdx.z / nh;
    const int hi = blockIdx.z % nh;
    A += (long)bi * sAb + (long)hi * sAh;
    B += (long)bi * sBb + (long)hi * sBh;
    const long coff = (long)bi * sCb + (long)hi * sCh;
    if (Cf) Cf += coff;
    if (Cb) Cb += coff;

    const int m0   = blockIdx.y * BM;
    const int n0   = blockIdx.x * BN;
    const int tid  = threadIdx.x;
    const int lane = tid & 31;
    const int w    = tid >> 5;   // wave id 0..7
    const int wm   = w & 3;      // 4 waves across M (4*32 = 128)
    const int wn   = w >> 2;     // 2 waves across N (2*32 = 64)

    // ---- register staging buffers (software pipeline) ----
    uint2  aReg[4];
    uint2  bVec[2];      // CONTIG path: 64x32 tile = 512 uint2
    __bf16 bReg[8];      // generic-stride path (QR^T)

    auto loadA = [&](int k0) {
        #pragma unroll
        for (int i = 0; i < 4; ++i) {           // 128x32 bf16 = 1024 uint2
            int p = tid + i * WG;
            int r = p >> 3;                     // 8 uint2 per row
            int c = (p & 7) << 2;
            aReg[i] = *(const uint2*)(A + (long)(m0 + r) * K + k0 + c);
        }
    };
    auto loadB = [&](int k0) {
        if (CONTIG) {
            #pragma unroll
            for (int i = 0; i < 2; ++i) {       // coalesced 8B loads
                int p = tid + i * WG;
                int k = p >> 4;                 // 16 uint2 per k-row
                int c = (p & 15) << 2;          // n offset within tile
                bVec[i] = *(const uint2*)(B + (long)(k0 + k) * sBk + n0 + c);
            }
        } else {
            #pragma unroll
            for (int i = 0; i < 8; ++i) {       // generic strides (QR^T)
                int q = tid + i * WG;
                int n = q >> 5;
                int k = q & 31;
                bReg[i] = B[(long)(k0 + k) * sBk + (long)(n0 + n) * sBn];
            }
        }
    };
    auto storeTiles = [&]() {
        #pragma unroll
        for (int i = 0; i < 4; ++i) {
            int p = tid + i * WG;
            int r = p >> 3;
            int c = (p & 7) << 2;
            *(uint2*)&As[r][c] = aReg[i];       // 8B-aligned ds_store_b64
        }
        if (CONTIG) {
            #pragma unroll
            for (int i = 0; i < 2; ++i) {       // transposed scatter to Bs[n][k]
                int p = tid + i * WG;
                int k = p >> 4;
                int c = (p & 15) << 2;
                union { uint2 d; __bf16 h[4]; } u;
                u.d = bVec[i];
                #pragma unroll
                for (int j = 0; j < 4; ++j)
                    Bs[c + j][k] = u.h[j];
            }
        } else {
            #pragma unroll
            for (int i = 0; i < 8; ++i) {
                int q = tid + i * WG;
                int n = q >> 5;
                int k = q & 31;
                Bs[n][k] = bReg[i];
            }
        }
    };

    v8f acc[2][2] = {{{}, {}}, {{}, {}}};

    loadA(0);
    loadB(0);

    for (int k0 = 0; k0 < K; k0 += BK) {
        storeTiles();
        __syncthreads();

        // issue next tile's global loads before compute (overlap with WMMA)
        if (k0 + BK < K) {
            loadA(k0 + BK);
            loadB(k0 + BK);
            if (k0 + 2 * BK < K)  // gfx1250 global_prefetch_b8 two tiles ahead
                __builtin_prefetch(A + (long)(m0 + (tid >> 3)) * K + k0 + 2 * BK, 0, 1);
        }

        // --- gather fragments per documented 16-bit WMMA layouts ---
        union Frag { v16bf v; unsigned u[8]; };
        Frag fa[2], fb[2];
        const int ml  = lane & 15;
        const int kb  = (lane >> 4) << 3;        // A: K base 0 / 8
        const int kb2 = (lane >> 4) << 4;        // unused; B also uses kb (0 / 8)
        const int mw  = wm * 32;
        const int nw  = wn * 32;
        #pragma unroll
        for (int j = 0; j < 8; ++j) {
            int offa = 2 * j + ((j & 4) << 1);   // {0,2,4,6,16,18,20,22}
            fa[0].u[j] = *(const unsigned*)&As[mw + ml][kb + offa];
            fa[1].u[j] = *(const unsigned*)&As[mw + 16 + ml][kb + offa];
            fb[0].u[j] = *(const unsigned*)&Bs[nw + ml][kb + offa];
            fb[1].u[j] = *(const unsigned*)&Bs[nw + 16 + ml][kb + offa];
        }
        #pragma unroll
        for (int sm = 0; sm < 2; ++sm)
            #pragma unroll
            for (int sn = 0; sn < 2; ++sn)
                acc[sm][sn] = __builtin_amdgcn_wmma_f32_16x16x32_bf16(
                    false, fa[sm].v, false, fb[sn].v, (short)0,
                    acc[sm][sn], false, false);
        __syncthreads();
    }

    // --- epilogue: D layout lane l -> N = l%16, VGPR e -> M = e + 8*(l/16) ---
    const int lm = (lane >> 4) << 3;
    const int ln = lane & 15;
    #pragma unroll
    for (int sm = 0; sm < 2; ++sm) {
        #pragma unroll
        for (int e = 0; e < 8; ++e) {
            int gm  = m0 + wm * 32 + sm * 16 + lm + e;
            int gn0 = n0 + wn * 32 + ln;
            int gn1 = gn0 + 16;
            float v0 = acc[sm][0][e], v1 = acc[sm][1][e];
            if (epi == EPI_RELU) {
                v0 = fmaxf(v0, 0.f); v1 = fmaxf(v1, 0.f);
            } else if (epi == EPI_MASKSCALE) {   // keep strict lower tri (s<t)
                v0 *= scale; v1 *= scale;
                if (gn0 >= gm) v0 = 0.f;
                if (gn1 >= gm) v1 = 0.f;
            }
            long o0 = (long)gm * Nd + gn0;
            long o1 = (long)gm * Nd + gn1;
            if (Cf) { Cf[o0] = v0; Cf[o1] = v1; }
            if (Cb) { Cb[o0] = (__bf16)v0; Cb[o1] = (__bf16)v1; }
        }
    }
}

// ---------------------------------------------------------------------------
// Row LayerNorm (no affine):  out = LN(x + a1 + a2), one block per row.
// In-place safe (values cached in registers before reduction).
// ---------------------------------------------------------------------------
__global__ __launch_bounds__(WG) void k_ln(
    const float* __restrict__ x, const float* __restrict__ a1,
    const float* __restrict__ a2, float* __restrict__ of,
    __bf16* __restrict__ ob, int ncols)
{
    const long row = blockIdx.x;
    const long base = row * (long)ncols;
    const int tid = threadIdx.x;

    float vals[8];
    int cnt = 0;
    float s = 0.f, s2 = 0.f;
    for (int c = tid; c < ncols; c += WG) {
        float v = x[base + c];
        if (a1) v += a1[base + c];
        if (a2) v += a2[base + c];
        vals[cnt++] = v; s += v; s2 += v * v;
    }
    __shared__ float rs[WG], rq[WG];
    rs[tid] = s; rq[tid] = s2;
    __syncthreads();
    for (int off = WG / 2; off > 0; off >>= 1) {
        if (tid < off) { rs[tid] += rs[tid + off]; rq[tid] += rq[tid + off]; }
        __syncthreads();
    }
    float mean = rs[0] / ncols;
    float var  = rq[0] / ncols - mean * mean;
    float inv  = rsqrtf(var + 1e-5f);
    cnt = 0;
    for (int c = tid; c < ncols; c += WG) {
        float o = (vals[cnt++] - mean) * inv;
        if (of) of[base + c] = o;
        if (ob) ob[base + c] = (__bf16)o;
    }
}

// inp[r,:] = LN(embed_w[idx[r],:])
__global__ __launch_bounds__(WG) void k_embed_ln(
    const int* __restrict__ idx, const float* __restrict__ emb,
    float* __restrict__ out, int D)
{
    const long row = blockIdx.x;
    const float* xr = emb + (long)idx[row] * D;
    const int tid = threadIdx.x;

    float vals[8];
    int cnt = 0;
    float s = 0.f, s2 = 0.f;
    for (int c = tid; c < D; c += WG) {
        float v = xr[c];
        vals[cnt++] = v; s += v; s2 += v * v;
    }
    __shared__ float rs[WG], rq[WG];
    rs[tid] = s; rq[tid] = s2;
    __syncthreads();
    for (int off = WG / 2; off > 0; off >>= 1) {
        if (tid < off) { rs[tid] += rs[tid + off]; rq[tid] += rq[tid + off]; }
        __syncthreads();
    }
    float mean = rs[0] / D;
    float inv  = rsqrtf(rq[0] / D - mean * mean + 1e-5f);
    cnt = 0;
    for (int c = tid; c < D; c += WG)
        out[row * (long)D + c] = (vals[cnt++] - mean) * inv;
}

// RoPE with even-quantized exponent + mod-1 phase wrap (matches reference).
__global__ __launch_bounds__(WG) void k_rope(
    const float* __restrict__ xs, __bf16* __restrict__ qr,
    long total, int NF, int T)
{
    for (long i = blockIdx.x * (long)blockDim.x + threadIdx.x; i < total;
         i += (long)gridDim.x * blockDim.x) {
        int n  = (int)(i % NF);
        int t  = (int)((i / NF) % T);
        float v  = xs[i];
        float vr = (n & 1) ? xs[i - 1] : -xs[i + 1];
        float q  = (float)(n & ~1);
        float fr = exp2f(-16.0f * q / (float)NF) * 0.15915494309189535f; // /(2pi)
        float ph = (float)t * fr;
        float ang = (ph - floorf(ph)) * 6.283185307179586f;
        qr[i] = (__bf16)(v * cosf(ang) + vr * sinf(ang));
    }
}

// xy[b,t,h*NF+n] = x_sparse[b,h,t,n] * y_sparse[b,h,t,n]  (bf16 out)
__global__ __launch_bounds__(WG) void k_mulpack(
    const float* __restrict__ xs, const float* __restrict__ ys,
    __bf16* __restrict__ xy, long total, int NF, int T, int NH)
{
    for (long i = blockIdx.x * (long)blockDim.x + threadIdx.x; i < total;
         i += (long)gridDim.x * blockDim.x) {
        int  n = (int)(i % NF);
        long r = i / NF;
        int  t = (int)(r % T);
        long r2 = r / T;
        int  h = (int)(r2 % NH);
        long b = r2 / NH;
        xy[(((b * T + t) * NH) + h) * (long)NF + n] = (__bf16)(xs[i] * ys[i]);
    }
}

__global__ __launch_bounds__(WG) void k_cvt(
    const float* __restrict__ s, __bf16* __restrict__ d, long n)
{
    for (long i = blockIdx.x * (long)blockDim.x + threadIdx.x; i < n;
         i += (long)gridDim.x * blockDim.x)
        d[i] = (__bf16)s[i];
}

__global__ __launch_bounds__(WG) void k_bcast(
    const float* __restrict__ init, float* __restrict__ dst, long n, int D)
{
    for (long i = blockIdx.x * (long)blockDim.x + threadIdx.x; i < n;
         i += (long)gridDim.x * blockDim.x)
        dst[i] = init[i % D];
}

// ---------------------------------------------------------------------------
// Host orchestration
// ---------------------------------------------------------------------------
extern "C" void kernel_launch(void* const* d_in, const int* in_sizes, int n_in,
                              void* d_out, int out_size, void* d_ws, size_t ws_size,
                              hipStream_t stream)
{
    (void)in_sizes; (void)n_in; (void)out_size; (void)ws_size;

    const int  Bb = 4, T = 512, D = 512, NH = 8, NF = 4096, V = 8192;
    const long BT = (long)Bb * T;

    const int*   idx      = (const int*)  d_in[0];
    const float* embed_w  = (const float*)d_in[1];
    const float* wsrc[6]  = { (const float*)d_in[2], (const float*)d_in[3],
                              (const float*)d_in[4], (const float*)d_in[5],
                              (const float*)d_in[6], (const float*)d_in[7] };
    const float* lm_head  = (const float*)d_in[8];
    const float* h_init   = (const float*)d_in[9];
    const float* l_init   = (const float*)d_in[10];

    // bump allocator over workspace
    char*  base = (char*)d_ws;
    size_t off  = 0;
    auto alloc = [&](size_t bytes) -> char* {
        size_t a = (off + 255) & ~(size_t)255;
        off = a + bytes;
        return base + a;
    };

    const long WSZ = (long)NH * D * NF;           // per weight tensor
    __bf16* wb[6];
    for (int i = 0; i < 6; ++i) wb[i] = (__bf16*)alloc(sizeof(__bf16) * WSZ);
    __bf16* wlm = (__bf16*)alloc(sizeof(__bf16) * (long)D * V);

    float*  inp   = (float*) alloc(4 * BT * D);
    float*  zH    = (float*) alloc(4 * BT * D);
    float*  zL    = (float*) alloc(4 * BT * D);
    float*  xi    = (float*) alloc(4 * BT * D);
    float*  ypre  = (float*) alloc(4 * BT * D);
    float*  y2    = (float*) alloc(4 * BT * D);
    float*  attn  = (float*) alloc(4 * (long)Bb * NH * T * D);
    float*  xs    = (float*) alloc(4 * (long)Bb * NH * T * NF);
    float*  ys    = (float*) alloc(4 * (long)Bb * NH * T * NF);
    __bf16* xi_b  = (__bf16*)alloc(2 * BT * D);
    __bf16* QR    = (__bf16*)alloc(2 * (long)Bb * NH * T * NF);
    __bf16* sc    = (__bf16*)alloc(2 * (long)Bb * NH * T * T);
    __bf16* yKVb  = (__bf16*)alloc(2 * (long)Bb * NH * T * D);
    __bf16* xy    = (__bf16*)alloc(2 * (long)Bb * NH * T * NF);
    __bf16* zH_b  = (__bf16*)alloc(2 * BT * D);

    auto gemm = [&](const __bf16* A, long sAb, long sAh,
                    const __bf16* Bp, long sBb, long sBh, long sBk, long sBn,
                    float* Cf, __bf16* Cbf, long sCb, long sCh,
                    int M, int Nd, int K, int batch, int epi, float scale) {
        dim3 g(Nd / BN, M / BM, batch);
        if (sBn == 1)
            k_gemm<true><<<g, WG, 0, stream>>>(A, sAb, sAh, Bp, sBb, sBh, sBk, sBn,
                                               Cf, Cbf, sCb, sCh, M, Nd, K, NH, epi, scale);
        else
            k_gemm<false><<<g, WG, 0, stream>>>(A, sAb, sAh, Bp, sBb, sBh, sBk, sBn,
                                                Cf, Cbf, sCb, sCh, M, Nd, K, NH, epi, scale);
    };

    // weights -> bf16 (every call: deterministic, graph-capture safe)
    for (int i = 0; i < 6; ++i)
        k_cvt<<<4096, WG, 0, stream>>>(wsrc[i], wb[i], WSZ);
    k_cvt<<<2048, WG, 0, stream>>>(lm_head, wlm, (long)D * V);

    k_embed_ln<<<(unsigned)BT, WG, 0, stream>>>(idx, embed_w, inp, D);
    k_bcast<<<2048, WG, 0, stream>>>(h_init, zH, BT * D, D);
    k_bcast<<<2048, WG, 0, stream>>>(l_init, zL, BT * D, D);

    const long tot = (long)Bb * NH * T * NF;

    auto block = [&](float* x, const float* injA, const float* injB,
                     const __bf16* enc, const __bf16* encv, const __bf16* dec) {
        // xi = LN(x + injA (+ injB))
        k_ln<<<(unsigned)BT, WG, 0, stream>>>(x, injA, injB, xi, xi_b, D);
        // x_sparse = relu(xi @ enc[h])          (T x D)(D x NF) per (b,h)
        gemm(xi_b, (long)T * D, 0,
             enc, 0, (long)D * NF, NF, 1,
             xs, nullptr, (long)NH * T * NF, (long)T * NF,
             T, NF, D, Bb * NH, EPI_RELU, 1.f);
        // QR = rope(x_sparse)
        k_rope<<<4096, WG, 0, stream>>>(xs, QR, tot, NF, T);
        // scores = mask(QR @ QR^T * N^-0.5)     (T x NF)(NF x T) per (b,h)
        gemm(QR, (long)NH * T * NF, (long)T * NF,
             QR, (long)NH * T * NF, (long)T * NF, 1, NF,
             nullptr, sc, (long)NH * T * T, (long)T * T,
             T, T, NF, Bb * NH, EPI_MASKSCALE, 1.0f / 64.0f);
        // attn = scores @ xi                    (T x T)(T x D) per (b,h)
        gemm(sc, (long)NH * T * T, (long)T * T,
             xi_b, (long)T * D, 0, D, 1,
             attn, nullptr, (long)NH * T * D, (long)T * D,
             T, D, T, Bb * NH, EPI_NONE, 1.f);
        // yKV = LN(attn) over D
        k_ln<<<(unsigned)(Bb * NH * T), WG, 0, stream>>>(
            attn, nullptr, nullptr, nullptr, yKVb, D);
        // y_sparse = relu(yKV @ enc_v[h])
        gemm(yKVb, (long)NH * T * D, (long)T * D,
             encv, 0, (long)D * NF, NF, 1,
             ys, nullptr, (long)NH * T * NF, (long)T * NF,
             T, NF, D, Bb * NH, EPI_RELU, 1.f);
        // xy = (x_sparse * y_sparse) packed to (BT, NH*NF)
        k_mulpack<<<4096, WG, 0, stream>>>(xs, ys, xy, tot, NF, T, NH);
        // ypre = xy @ dec                        (BT x NH*NF)(NH*NF x D)
        gemm(xy, 0, 0, dec, 0, 0, D, 1,
             ypre, nullptr, 0, 0, (int)BT, D, NH * NF, 1, EPI_NONE, 1.f);
        // y = LN(ypre); x = LN(x + y)
        k_ln<<<(unsigned)BT, WG, 0, stream>>>(ypre, nullptr, nullptr, y2, nullptr, D);
        k_ln<<<(unsigned)BT, WG, 0, stream>>>(x, y2, nullptr, x, nullptr, D);
    };

    const __bf16 *henc = wb[0], *hencv = wb[1], *hdec = wb[2];
    const __bf16 *lenc = wb[3], *lencv = wb[4], *ldec = wb[5];

    for (int h = 0; h < 3; ++h) {
        for (int l = 0; l < 2; ++l)
            if (!(h == 2 && l == 1))
                block(zL, zH, inp, lenc, lencv, ldec);
        if (h < 2)
            block(zH, zL, nullptr, henc, hencv, hdec);
    }
    block(zL, zH, inp, lenc, lencv, ldec);
    block(zH, zL, nullptr, henc, hencv, hdec);

    // logits = z_H @ lm_head -> d_out (f32)
    k_cvt<<<2048, WG, 0, stream>>>(zH, zH_b, BT * D);
    gemm(zH_b, 0, 0, wlm, 0, 0, V, 1,
         (float*)d_out, nullptr, 0, 0, (int)BT, V, D, 1, EPI_NONE, 1.f);
}
